// PokeBNN_27066883899828
// MI455X (gfx1250) — 8-bit matrix operand order repaired and checked by reading, compiled, not run on hardware
//
#include <hip/hip_runtime.h>
#include <stdint.h>

typedef __attribute__((ext_vector_type(8))) int v8i;

#define BATCH 8

__device__ __forceinline__ void async_ld_b128(void* lds, const void* g) {
#if defined(__gfx1250__)
  unsigned l = (unsigned)(uintptr_t)lds;
  unsigned long long ga = (unsigned long long)(uintptr_t)g;
  asm volatile("global_load_async_to_lds_b128 %0, %1, off"
               :: "v"(l), "v"(ga) : "memory");
#else
  (void)lds; (void)g;
#endif
}

__device__ __forceinline__ void wait_async0() {
#if defined(__gfx1250__)
  asm volatile("s_wait_asynccnt 0x0" ::: "memory");
#endif
}

__global__ __launch_bounds__(256) void k_absmax(const float* __restrict__ x, int n,
                                                unsigned* __restrict__ out) {
  __shared__ float smem[8];
  float m = 0.f;
  for (int i = blockIdx.x * blockDim.x + threadIdx.x; i < n; i += gridDim.x * blockDim.x)
    m = fmaxf(m, fabsf(x[i]));
  for (int off = 16; off > 0; off >>= 1) m = fmaxf(m, __shfl_down(m, off, 32));
  int lane = threadIdx.x & 31, wid = threadIdx.x >> 5;
  if (lane == 0) smem[wid] = m;
  __syncthreads();
  if (wid == 0) {
    m = (lane < 8) ? smem[lane] : 0.f;
    for (int off = 4; off > 0; off >>= 1) m = fmaxf(m, __shfl_down(m, off, 32));
    if (lane == 0) atomicMax(out, __float_as_uint(m));
  }
}

__device__ __forceinline__ float scale_from(const unsigned* a, float denom) {
  return fmaxf(__uint_as_float(*a), 1e-6f) / denom;
}

__global__ __launch_bounds__(256) void k_quant_i8(const float* __restrict__ x,
                                                  const unsigned* __restrict__ amax,
                                                  int8_t* __restrict__ q, int n) {
  float inv = 1.f / scale_from(amax, 127.f);
  for (int i = blockIdx.x * blockDim.x + threadIdx.x; i < n; i += gridDim.x * blockDim.x) {
    float v = rintf(x[i] * inv);
    v = fminf(fmaxf(v, -127.f), 127.f);
    q[i] = (int8_t)(int)v;
  }
}

__global__ __launch_bounds__(256) void k_fakequant(const float* __restrict__ x,
                                                   const unsigned* __restrict__ amax,
                                                   float* __restrict__ y, int n) {
  float s = scale_from(amax, 127.f);
  for (int i = blockIdx.x * blockDim.x + threadIdx.x; i < n; i += gridDim.x * blockDim.x)
    y[i] = fminf(fmaxf(rintf(x[i] / s), -127.f), 127.f) * s;
}

__global__ __launch_bounds__(256) void k_im2col_i8(const int8_t* __restrict__ x,
    int B, int H, int W, int C, int KH, int KW, int SH, int SW, int PH, int PW,
    int HO, int WO, int K, int Kpad, int8_t* __restrict__ A) {
  long total = (long)B * HO * WO * Kpad;
  for (long i = blockIdx.x * (long)blockDim.x + threadIdx.x; i < total;
       i += (long)gridDim.x * blockDim.x) {
    int k = (int)(i % Kpad);
    long m = i / Kpad;
    int ow = (int)(m % WO); long t = m / WO;
    int oh = (int)(t % HO); int b = (int)(t / HO);
    int8_t v = 0;
    if (k < K) {
      int c = k % C; int kk = k / C; int kw = kk % KW; int kh = kk / KW;
      int ih = oh * SH + kh - PH, iw = ow * SW + kw - PW;
      if (ih >= 0 && ih < H && iw >= 0 && iw < W)
        v = x[(((long)b * H + ih) * W + iw) * C + c];
    }
    A[i] = v;
  }
}

__global__ __launch_bounds__(256) void k_quant_pack_w(const float* __restrict__ w,
    const unsigned* __restrict__ amax, int K, int N, int Kpad, int Npad,
    int8_t* __restrict__ Bt) {
  float inv = 1.f / scale_from(amax, 127.f);
  long total = (long)Npad * Kpad;
  for (long i = blockIdx.x * (long)blockDim.x + threadIdx.x; i < total;
       i += (long)gridDim.x * blockDim.x) {
    int k = (int)(i % Kpad); int n = (int)(i / Kpad);
    int8_t q = 0;
    if (k < K && n < N) {
      float v = rintf(w[(long)k * N + n] * inv);
      q = (int8_t)(int)fminf(fmaxf(v, -127.f), 127.f);
    }
    Bt[i] = q;
  }
}

__global__ void k_bn_coeffs(const float* __restrict__ scale, const float* __restrict__ bias,
                            const float* __restrict__ mean, const float* __restrict__ var,
                            int C, float* __restrict__ A, float* __restrict__ B) {
  int c = blockIdx.x * blockDim.x + threadIdx.x;
  if (c < C) {
    float a = scale[c] / sqrtf(var[c] + 1e-5f);
    A[c] = a;
    B[c] = bias[c] - mean[c] * a;
  }
}

#define BM 128
#define BN 64
#define BK 64

__global__ __launch_bounds__(256) void k_gemm_i8_wmma(const int8_t* __restrict__ A,
    const int8_t* __restrict__ Bt, int M, int N, int Npad, int Kpad,
    const unsigned* __restrict__ amaxA, const unsigned* __restrict__ amaxW,
    const float* __restrict__ bnA, const float* __restrict__ bnB,
    float* __restrict__ C) {
  __shared__ __align__(16) int8_t As[2][BM * BK];
  __shared__ __align__(16) int8_t Bs[2][BN * BK];
  const int tid = threadIdx.x;
  const int wid = tid >> 5, lane = tid & 31;
  const int wm = wid >> 1, wn = wid & 1;
  const int m0 = blockIdx.x * BM, n0 = blockIdx.y * BN;
  const int ml = lane & 15, h = lane >> 4;

  v8i acc[2][2];
#pragma unroll
  for (int i = 0; i < 2; ++i)
#pragma unroll
    for (int j = 0; j < 2; ++j) acc[i][j] = (v8i)0;

  auto stage = [&](int k0, int buf) {
#pragma unroll
    for (int it = 0; it < 2; ++it) {
      int idx = tid + it * 256;
      int r = idx >> 2, cb = (idx & 3) << 4;
      int gm = m0 + r;
      int8_t* dst = &As[buf][r * BK + cb];
      if (gm < M) async_ld_b128(dst, A + (long)gm * Kpad + k0 + cb);
      else        *(uint4*)dst = make_uint4(0u, 0u, 0u, 0u);
    }
    {
      int r = tid >> 2, cb = (tid & 3) << 4;
      int gn = n0 + r;
      int8_t* dst = &Bs[buf][r * BK + cb];
      if (gn < Npad) async_ld_b128(dst, Bt + (long)gn * Kpad + k0 + cb);
      else           *(uint4*)dst = make_uint4(0u, 0u, 0u, 0u);
    }
  };

  const int nT = Kpad / BK;
  stage(0, 0);
  wait_async0();
  __syncthreads();

  for (int t = 0; t < nT; ++t) {
    const int cur = t & 1;
    if (t + 1 < nT) stage((t + 1) * BK, cur ^ 1);

    v8i afrag[2], bfrag[2];
#pragma unroll
    for (int sm = 0; sm < 2; ++sm) {
      int row = (wm << 5) + (sm << 4) + ml;
      const int8_t* ap = &As[cur][row * BK];
#pragma unroll
      for (int j = 0; j < 8; ++j) {
        int ks = ((j >> 1) << 4) + (h << 3) + ((j & 1) << 2);
        afrag[sm][j] = *(const int*)(ap + ks);
      }
    }
#pragma unroll
    for (int sn = 0; sn < 2; ++sn) {
      int nrow = (wn << 5) + (sn << 4) + ml;
      const int8_t* bp = &Bs[cur][nrow * BK];
#pragma unroll
      for (int j = 0; j < 8; ++j) {
        int ks = ((j >> 1) << 4) + (h << 3) + ((j & 1) << 2);
        bfrag[sn][j] = *(const int*)(bp + ks);
      }
    }
#pragma unroll
    for (int sm = 0; sm < 2; ++sm)
#pragma unroll
      for (int sn = 0; sn < 2; ++sn)
        acc[sm][sn] = __builtin_amdgcn_wmma_i32_16x16x64_iu8(
            true, afrag[sm], true, bfrag[sn], acc[sm][sn], false, false);

    if (t + 1 < nT) wait_async0();
    __syncthreads();
  }

  __builtin_amdgcn_s_wait_tensorcnt(0);

  const float sc = scale_from(amaxA, 127.f) * scale_from(amaxW, 127.f);
#pragma unroll
  for (int sm = 0; sm < 2; ++sm)
#pragma unroll
    for (int sn = 0; sn < 2; ++sn)
#pragma unroll
      for (int v = 0; v < 8; ++v) {
        int gm = m0 + (wm << 5) + (sm << 4) + (h << 3) + v;
        int gn = n0 + (wn << 5) + (sn << 4) + ml;
        if (gm < M && gn < N) {
          float y = (float)acc[sm][sn][v] * sc;
          if (bnA) y = y * bnA[gn] + bnB[gn];
          C[(long)gm * N + gn] = y;
        }
      }
}

__device__ __forceinline__ float chmap(const float* __restrict__ r, long base, int c,
                                       int Cin, int Cout, int num, int tileMode) {
  if (Cin > Cout) {
    float s = 0.f;
    for (int j = 0; j < num; ++j) s += r[base + c * num + j];
    return s / (float)num;
  } else if (Cin < Cout) {
    if (tileMode) return r[base + (c % Cin)];
    return (c < Cin) ? r[base + c] : 0.f;
  }
  return r[base + c];
}

__global__ __launch_bounds__(256) void k_reshape_add(float* __restrict__ y,
    const float* __restrict__ r, int B, int HO, int WO, int Cout,
    int H, int W, int Cin, int tileMode) {
  long total = (long)B * HO * WO * Cout;
  bool pool = (H != HO) || (W != WO);
  int num = (Cin > Cout) ? (Cin / Cout) : 1;
  for (long i = blockIdx.x * (long)blockDim.x + threadIdx.x; i < total;
       i += (long)gridDim.x * blockDim.x) {
    int c = (int)(i % Cout); long t = i / Cout;
    int ow = (int)(t % WO); t /= WO;
    int oh = (int)(t % HO); int b = (int)(t / HO);
    float rv;
    if (!pool) {
      long base = (((long)b * H + oh) * W + ow) * Cin;
      rv = chmap(r, base, c, Cin, Cout, num, tileMode);
    } else {
      float s = 0.f;
      for (int kh = 0; kh < 3; ++kh)
        for (int kw = 0; kw < 3; ++kw) {
          int ih = oh * 2 + kh, iw = ow * 2 + kw;
          if (ih < H && iw < W) {
            long base = (((long)b * H + ih) * W + iw) * Cin;
            s += chmap(r, base, c, Cin, Cout, num, tileMode);
          }
        }
      rv = s / 9.f;
    }
    y[i] += rv;
  }
}

__global__ __launch_bounds__(256) void k_dprelu_se_bn(float* __restrict__ y,
    const float* __restrict__ bx, const float* __restrict__ by,
    const float* __restrict__ ns, const float* __restrict__ ps,
    const float* __restrict__ se, const float* __restrict__ bnA,
    const float* __restrict__ bnB, int B, int HW, int C) {
  long total = (long)B * HW * C;
  for (long i = blockIdx.x * (long)blockDim.x + threadIdx.x; i < total;
       i += (long)gridDim.x * blockDim.x) {
    int c = (int)(i % C);
    int b = (int)(i / ((long)HW * C));
    float v = y[i];
    float xc = v - bx[c];
    float d = ((xc >= 0.f) ? ps[c] : ns[c]) * xc - by[c];
    if (se) d *= se[b * C + c];
    if (bnA) d = d * bnA[c] + bnB[c];
    y[i] = d;
  }
}

__global__ __launch_bounds__(256) void k_spatial_mean(const float* __restrict__ x,
    float* __restrict__ out, int B, int HW, int C) {
  int total = B * C;
  for (int i = blockIdx.x * blockDim.x + threadIdx.x; i < total; i += gridDim.x * blockDim.x) {
    int c = i % C, b = i / C;
    const float* p = x + (long)b * HW * C + c;
    float s = 0.f;
    for (int t = 0; t < HW; ++t) s += p[(long)t * C];
    out[i] = s / (float)HW;
  }
}

__global__ __launch_bounds__(256) void k_se_mm1(const float* __restrict__ s,
    const float* __restrict__ w1, const float* __restrict__ b1,
    const unsigned* amaxS, const unsigned* amaxW, float* __restrict__ o,
    int B, int Cin, int P) {
  float ss = scale_from(amaxS, 127.f), sw = scale_from(amaxW, 127.f);
  int total = B * P;
  for (int i = blockIdx.x * blockDim.x + threadIdx.x; i < total; i += gridDim.x * blockDim.x) {
    int j = i % P, b = i / P;
    float acc = 0.f;
    for (int c = 0; c < Cin; ++c) {
      float a = fminf(fmaxf(rintf(s[b * Cin + c] / ss), -127.f), 127.f) * ss;
      float w = fminf(fmaxf(rintf(w1[(long)c * P + j] / sw), -127.f), 127.f) * sw;
      acc += a * w;
    }
    o[i] = fmaxf(acc + b1[j], 0.f);
  }
}

__global__ __launch_bounds__(256) void k_se_mm2(const float* __restrict__ h,
    const float* __restrict__ w2, const float* __restrict__ b2,
    const unsigned* amaxH, const unsigned* amaxW, float* __restrict__ seb,
    int B, int P, int Cout) {
  float sh = scale_from(amaxH, 255.f), sw = scale_from(amaxW, 127.f);
  int total = B * Cout;
  for (int i = blockIdx.x * blockDim.x + threadIdx.x; i < total; i += gridDim.x * blockDim.x) {
    int co = i % Cout, b = i / Cout;
    float acc = 0.f;
    for (int j = 0; j < P; ++j) {
      float a = fminf(fmaxf(rintf(h[b * P + j] / sh), 0.f), 255.f) * sh;
      float w = fminf(fmaxf(rintf(w2[(long)j * Cout + co] / sw), -127.f), 127.f) * sw;
      acc += a * w;
    }
    acc += b2[co];
    seb[i] = fminf(fmaxf(acc + 3.f, 0.f), 6.f) / 6.f;
  }
}

__global__ __launch_bounds__(256) void k_groupconv3x3(const float* __restrict__ x,
    const float* __restrict__ w, const float* __restrict__ bnA,
    const float* __restrict__ bnB, float* __restrict__ y,
    int B, int H, int W, int Cin, int Cout) {
  long total = (long)B * H * W * Cout;
  int per = Cout / Cin;
  for (long i = blockIdx.x * (long)blockDim.x + threadIdx.x; i < total;
       i += (long)gridDim.x * blockDim.x) {
    int o = (int)(i % Cout); long t = i / Cout;
    int iw = (int)(t % W); t /= W;
    int ih = (int)(t % H); int b = (int)(t / H);
    int g = o / per;
    float acc = 0.f;
    for (int kh = 0; kh < 3; ++kh)
      for (int kw = 0; kw < 3; ++kw) {
        int y2 = ih + kh - 1, x2 = iw + kw - 1;
        if (y2 >= 0 && y2 < H && x2 >= 0 && x2 < W)
          acc += x[(((long)b * H + y2) * W + x2) * Cin + g] * w[(kh * 3 + kw) * Cout + o];
      }
    y[i] = acc * bnA[o] + bnB[o];
  }
}

static inline int grid_for(long n) {
  long g = (n + 255) / 256;
  if (g > 4096) g = 4096;
  if (g < 1) g = 1;
  return (int)g;
}

extern "C" void kernel_launch(void* const* d_in, const int* in_sizes, int n_in,
                              void* d_out, int out_size, void* d_ws, size_t ws_size,
                              hipStream_t stream) {
  (void)in_sizes; (void)n_in; (void)out_size; (void)ws_size;

  auto F = [&](int i) { return (const float*)d_in[i]; };
  const int LX = 0, L_ICW = 1, L_IBN = 2, L_IDPR = 6, L_IGW = 10, L_IBNG = 11,
            L_IDPRG = 15, L_BLK = 19, L_DENSE = 19 + 16 * 51;

  size_t off = 0;
  auto alloc = [&](size_t bytes) -> char* {
    off = (off + 255) & ~(size_t)255;
    char* p = (char*)d_ws + off;
    off += bytes;
    return p;
  };
  const size_t ACT_BYTES = (size_t)BATCH * 56 * 56 * 256 * sizeof(float);
  unsigned* scaleBase = (unsigned*)alloc(4096);
  float* slots[3];
  for (int i = 0; i < 3; ++i) slots[i] = (float*)alloc(ACT_BYTES);
  int8_t* qact   = (int8_t*)alloc((size_t)8 << 20);
  int8_t* im2col = (int8_t*)alloc((size_t)16 << 20);
  int8_t* wpack  = (int8_t*)alloc((size_t)8 << 20);
  float* bnA  = (float*)alloc(2048 * 4);
  float* bnB  = (float*)alloc(2048 * 4);
  float* bnA2 = (float*)alloc(2048 * 4);
  float* bnB2 = (float*)alloc(2048 * 4);
  float* seS  = (float*)alloc((size_t)BATCH * 2048 * 4);
  float* seH  = (float*)alloc((size_t)BATCH * 256 * 4);
  float* seB  = (float*)alloc((size_t)BATCH * 2048 * 4);
  float* pm   = (float*)alloc((size_t)BATCH * 2048 * 4);
  float* gwf  = (float*)alloc(3 * 3 * 64 * 4);

  hipMemsetAsync(scaleBase, 0, 4096, stream);
  int scaleCnt = 0;
  auto nsc = [&]() { return scaleBase + (scaleCnt++); };

  auto absmax = [&](const float* p, long n, unsigned* s) {
    k_absmax<<<grid_for(n), 256, 0, stream>>>(p, (int)n, s);
  };

  auto conv_wmma = [&](const float* Xin, float* Yout, const float* w,
                       const float* bns, const float* bnb, const float* bnm,
                       const float* bnv, int Cin, int H, int W, int Cout,
                       int KH, int S, int HO, int WO) {
    long nx = (long)BATCH * H * W * Cin;
    unsigned* sa = nsc();
    absmax(Xin, nx, sa);
    k_quant_i8<<<grid_for(nx), 256, 0, stream>>>(Xin, sa, qact, (int)nx);
    int M = BATCH * HO * WO, K = KH * KH * Cin;
    int Kpad = ((K + 63) / 64) * 64, Npad = ((Cout + 15) / 16) * 16;
    const int8_t* Amat = qact;
    if (KH != 1 || S != 1) {
      int padT = (HO - 1) * S + KH - H; if (padT < 0) padT = 0;
      int PH = padT / 2;
      long tot = (long)M * Kpad;
      k_im2col_i8<<<grid_for(tot), 256, 0, stream>>>(qact, BATCH, H, W, Cin, KH, KH,
                                                     S, S, PH, PH, HO, WO, K, Kpad, im2col);
      Amat = im2col;
    }
    unsigned* sw = nsc();
    absmax(w, (long)K * Cout, sw);
    k_quant_pack_w<<<grid_for((long)Npad * Kpad), 256, 0, stream>>>(w, sw, K, Cout,
                                                                    Kpad, Npad, wpack);
    k_bn_coeffs<<<(Cout + 255) / 256, 256, 0, stream>>>(bns, bnb, bnm, bnv, Cout, bnA, bnB);
    dim3 g((M + BM - 1) / BM, (Cout + BN - 1) / BN);
    k_gemm_i8_wmma<<<g, 256, 0, stream>>>(Amat, wpack, M, Cout, Npad, Kpad, sa, sw,
                                          bnA, bnB, Yout);
  };

  auto poke = [&](const float* Xin, const float* R1, float* Yout, int base,
                  int Cin, int H, int W, int Cout, int KH, int S,
                  int Cr1, int Hr1, int Wr1) {
    const float* w    = F(base + 0);
    const float* bns  = F(base + 1), *bnb = F(base + 2), *bnm = F(base + 3), *bnv = F(base + 4);
    const float* dbx  = F(base + 5), *dby = F(base + 6), *dns = F(base + 7), *dps = F(base + 8);
    const float* sew1 = F(base + 9), *seb1 = F(base + 10);
    const float* sew2 = F(base + 11), *seb2 = F(base + 12);
    const float* bss  = F(base + 13), *bsb = F(base + 14), *bsm = F(base + 15), *bsv = F(base + 16);
    int HO = (H + S - 1) / S, WO = (W + S - 1) / S;
    conv_wmma(Xin, Yout, w, bns, bnb, bnm, bnv, Cin, H, W, Cout, KH, S, HO, WO);
    long ny = (long)BATCH * HO * WO * Cout;
    k_reshape_add<<<grid_for(ny), 256, 0, stream>>>(Yout, Xin, BATCH, HO, WO, Cout,
                                                    H, W, Cin, 0);
    if (R1)
      k_reshape_add<<<grid_for(ny), 256, 0, stream>>>(Yout, R1, BATCH, HO, WO, Cout,
                                                      Hr1, Wr1, Cr1, 1);
    int P = Cin / 8;
    k_spatial_mean<<<grid_for(BATCH * Cin), 256, 0, stream>>>(Xin, seS, BATCH, H * W, Cin);
    unsigned* s1 = nsc(); absmax(seS, (long)BATCH * Cin, s1);
    unsigned* q1 = nsc(); absmax(sew1, (long)Cin * P, q1);
    k_se_mm1<<<grid_for(BATCH * P), 256, 0, stream>>>(seS, sew1, seb1, s1, q1, seH,
                                                      BATCH, Cin, P);
    unsigned* s2 = nsc(); absmax(seH, (long)BATCH * P, s2);
    unsigned* q2 = nsc(); absmax(sew2, (long)P * Cout, q2);
    k_se_mm2<<<grid_for(BATCH * Cout), 256, 0, stream>>>(seH, sew2, seb2, s2, q2, seB,
                                                         BATCH, P, Cout);
    k_bn_coeffs<<<(Cout + 255) / 256, 256, 0, stream>>>(bss, bsb, bsm, bsv, Cout, bnA2, bnB2);
    k_dprelu_se_bn<<<grid_for(ny), 256, 0, stream>>>(Yout, dbx, dby, dns, dps, seB,
                                                     bnA2, bnB2, BATCH, HO * WO, Cout);
  };

  const float* x = F(LX);
  conv_wmma(x, slots[0], F(L_ICW), F(L_IBN), F(L_IBN + 1), F(L_IBN + 2), F(L_IBN + 3),
            3, 224, 224, 32, 4, 4, 56, 56);
  long n1 = (long)BATCH * 56 * 56 * 32;
  k_dprelu_se_bn<<<grid_for(n1), 256, 0, stream>>>(slots[0], F(L_IDPR), F(L_IDPR + 1),
      F(L_IDPR + 2), F(L_IDPR + 3), nullptr, nullptr, nullptr, BATCH, 56 * 56, 32);
  unsigned* sgx = nsc(); absmax(slots[0], n1, sgx);
  k_fakequant<<<grid_for(n1), 256, 0, stream>>>(slots[0], sgx, slots[1], (int)n1);
  unsigned* sgw = nsc(); absmax(F(L_IGW), 3 * 3 * 64, sgw);
  k_fakequant<<<grid_for(3 * 3 * 64), 256, 0, stream>>>(F(L_IGW), sgw, gwf, 3 * 3 * 64);
  k_bn_coeffs<<<1, 256, 0, stream>>>(F(L_IBNG), F(L_IBNG + 1), F(L_IBNG + 2),
                                     F(L_IBNG + 3), 64, bnA, bnB);
  long n2 = (long)BATCH * 56 * 56 * 64;
  k_groupconv3x3<<<grid_for(n2), 256, 0, stream>>>(slots[1], gwf, bnA, bnB, slots[2],
                                                   BATCH, 56, 56, 32, 64);
  k_dprelu_se_bn<<<grid_for(n2), 256, 0, stream>>>(slots[2], F(L_IDPRG), F(L_IDPRG + 1),
      F(L_IDPRG + 2), F(L_IDPRG + 3), nullptr, nullptr, nullptr, BATCH, 56 * 56, 64);

  static const int FEAT[16] = {64, 64, 64, 128, 128, 128, 128, 256, 256, 256,
                               256, 256, 256, 512, 512, 512};
  static const int STR[16] = {1, 1, 1, 2, 1, 1, 1, 2, 1, 1, 1, 1, 1, 2, 1, 1};
  int xi = 2;
  int Cc = 64, Hc = 56;
  for (int b = 0; b < 16; ++b) {
    int f = FEAT[b], S = STR[b];
    int o1 = (xi + 1) % 3, o2 = (xi + 2) % 3;
    int base = L_BLK + b * 51;
    int HO = (Hc + S - 1) / S;
    poke(slots[xi], nullptr, slots[o1], base + 0, Cc, Hc, Hc, f, 1, 1, 0, 0, 0);
    poke(slots[o1], nullptr, slots[o2], base + 17, f, Hc, Hc, f, 3, S, 0, 0, 0);
    poke(slots[o2], slots[xi], slots[o1], base + 34, f, HO, HO, 4 * f, 1, 1, Cc, Hc, Hc);
    xi = o1;
    Cc = 4 * f;
    Hc = HO;
  }

  k_spatial_mean<<<grid_for(BATCH * Cc), 256, 0, stream>>>(slots[xi], pm, BATCH,
                                                           Hc * Hc, Cc);
  long np = (long)BATCH * Cc;
  unsigned* sa = nsc(); absmax(pm, np, sa);
  k_quant_i8<<<grid_for(np), 256, 0, stream>>>(pm, sa, qact, (int)np);
  unsigned* sw = nsc(); absmax(F(L_DENSE), (long)Cc * 1000, sw);
  int Npad = 1008;
  k_quant_pack_w<<<grid_for((long)Npad * Cc), 256, 0, stream>>>(F(L_DENSE), sw, Cc, 1000,
                                                                Cc, Npad, wpack);
  dim3 g((BATCH + BM - 1) / BM, (1000 + BN - 1) / BN);
  k_gemm_i8_wmma<<<g, 256, 0, stream>>>(qact, wpack, BATCH, 1000, Npad, Cc, sa, sw,
                                        nullptr, nullptr, (float*)d_out);
}
